// TopKMoE_77549929496706
// MI455X (gfx1250) — matrix operand order repaired and checked by reading, not compiled, not run on hardware
//
#include <hip/hip_runtime.h>
#include <math.h>
#include <stdint.h>

#define NTOK 8192   // B*T
#define DM   1024   // model dim
#define NE   8      // experts
#define NF   4096   // ffn dim

typedef __bf16 v16bf __attribute__((ext_vector_type(16)));
typedef float  v8f   __attribute__((ext_vector_type(8)));

union Frag { v16bf v; uint32_t u[8]; };

__device__ __forceinline__ uint16_t f2bf(float f) {
    uint32_t u = __float_as_uint(f);
    u += 0x7FFFu + ((u >> 16) & 1u);   // round-to-nearest-even
    return (uint16_t)(u >> 16);
}

__device__ __forceinline__ float gelu_exact(float v) {
    return 0.5f * v * (1.0f + erff(v * 0.70710678118654752f));
}

// ---- CDNA5 async copy: global -> LDS, 128b per lane, tracked by ASYNCcnt ----
__device__ __forceinline__ void async_b128(uint32_t lds_addr, const void* gaddr) {
    asm volatile("global_load_async_to_lds_b128 %0, %1, off"
                 :: "v"(lds_addr), "v"(gaddr) : "memory");
}

template <int N>
__device__ __forceinline__ void wait_asynccnt() {
#if __has_builtin(__builtin_amdgcn_s_wait_asynccnt)
    __builtin_amdgcn_s_wait_asynccnt((short)N);
#else
    asm volatile("s_wait_asynccnt %0" :: "n"(N) : "memory");
#endif
}

__device__ __forceinline__ uint32_t lds_addr32(const void* p) {
    // generic LDS address = {aperture_hi, lds_offset}; low 32 bits are the LDS byte offset
    return (uint32_t)(uintptr_t)p;
}

// ---------------- init: zero output + expert counts ----------------
__global__ void moe_init(float* __restrict__ out, int n, int* __restrict__ counts) {
    int i = blockIdx.x * blockDim.x + threadIdx.x;
    if (i < n) out[i] = 0.0f;
    if (i < NE) counts[i] = 0;
}

// ---------------- fp32 -> bf16, 8 elements per thread ----------------
__global__ void cvt_bf16v(const float* __restrict__ src, uint16_t* __restrict__ dst, int n) {
    int i = (blockIdx.x * blockDim.x + threadIdx.x) * 8;
    if (i + 8 > n) return;
    float4 f0 = *(const float4*)(src + i);
    float4 f1 = *(const float4*)(src + i + 4);
    union { uint4 q; uint16_t s[8]; } o;
    o.s[0] = f2bf(f0.x); o.s[1] = f2bf(f0.y); o.s[2] = f2bf(f0.z); o.s[3] = f2bf(f0.w);
    o.s[4] = f2bf(f1.x); o.s[5] = f2bf(f1.y); o.s[6] = f2bf(f1.z); o.s[7] = f2bf(f1.w);
    *(uint4*)(dst + i) = o.q;
}

// ---------------- gating: one wave32 per token ----------------
__global__ __launch_bounds__(256) void moe_gate(
    const float* __restrict__ x, const float* __restrict__ gw,
    int* __restrict__ counts, int* __restrict__ tok_ids, float* __restrict__ gate_vals)
{
    const int lane = threadIdx.x & 31;
    const int wid  = threadIdx.x >> 5;
    const int n    = blockIdx.x * 8 + wid;
    if (n >= NTOK) return;

    float acc[NE];
#pragma unroll
    for (int e = 0; e < NE; e++) acc[e] = 0.0f;

    const float* xr = x + (size_t)n * DM;
    for (int d = lane; d < DM; d += 32) {
        float xv = xr[d];
#pragma unroll
        for (int e = 0; e < NE; e++) acc[e] += xv * gw[d * NE + e];
    }
#pragma unroll
    for (int e = 0; e < NE; e++) {
#pragma unroll
        for (int off = 16; off > 0; off >>= 1)
            acc[e] += __shfl_xor(acc[e], off, 32);
    }
    if (lane == 0) {
        int   i0 = 0; float v0 = acc[0];
#pragma unroll
        for (int e = 1; e < NE; e++) if (acc[e] > v0) { v0 = acc[e]; i0 = e; }
        int   i1 = -1; float v1 = -3.0e38f;
#pragma unroll
        for (int e = 0; e < NE; e++) if (e != i0 && acc[e] > v1) { v1 = acc[e]; i1 = e; }
        float e1   = expf(v1 - v0);
        float norm = 1.0f / (1.0f + e1);

        int s0 = atomicAdd(&counts[i0], 1);
        tok_ids[i0 * NTOK + s0]   = n;
        gate_vals[i0 * NTOK + s0] = norm;
        int s1 = atomicAdd(&counts[i1], 1);
        tok_ids[i1 * NTOK + s1]   = n;
        gate_vals[i1 * NTOK + s1] = e1 * norm;
    }
}

// ---------------- GEMM1: h = gelu(gather(x) @ w1[e] + b1[e]) ----------------
// 128x128 block tile, K-step 32, double-buffered LDS, async A copies.
__global__ __launch_bounds__(256) void moe_gemm1(
    const uint16_t* __restrict__ xb, const uint16_t* __restrict__ wb1,
    const float* __restrict__ b1, const int* __restrict__ counts,
    const int* __restrict__ tok_ids, uint16_t* __restrict__ h, int e)
{
    __shared__ __align__(16) uint16_t As[2][128][32];   // [m][k]
    __shared__ __align__(16) uint16_t Bs[2][128][32];   // [n][k]

    const int cnt = counts[e];
    const int mt = blockIdx.x, nt = blockIdx.y;
    if (mt * 128 >= cnt) return;

    const int tid  = threadIdx.x;
    const int lane = tid & 31, wid = tid >> 5;
    const int wm   = wid & 3,  wn  = wid >> 2;

    const uint16_t* w1e = wb1 + (size_t)e * DM * NF;

    const int arow = tid >> 1, acol = (tid & 1) * 16;
    const int r_load = mt * 128 + arow;
    const int tok = (r_load < cnt) ? tok_ids[e * NTOK + r_load] : -1;
    const int bk = tid >> 3, bn = (tid & 7) * 16;

    const v8f vzero = {0,0,0,0,0,0,0,0};
    v8f acc[2][4];
#pragma unroll
    for (int i = 0; i < 2; i++)
#pragma unroll
        for (int j = 0; j < 4; j++) acc[i][j] = vzero;

    const int fm  = lane & 15;
    const int akk = (lane >> 4) * 8;
    const int bkk = (lane >> 4) * 8;

    auto stage = [&](int buf, int kt) {
        if (tok >= 0) {
            const uint16_t* src = xb + (size_t)tok * DM + kt + acol;
            uint32_t l = lds_addr32(&As[buf][arow][acol]);
            async_b128(l, src);
            async_b128(l + 16, src + 8);
        } else {
            const uint4 z = make_uint4(0u, 0u, 0u, 0u);
            *(uint4*)&As[buf][arow][acol]     = z;
            *(uint4*)&As[buf][arow][acol + 8] = z;
        }
        union { uint4 q[2]; uint16_t s[16]; } bv;
        const uint16_t* bsrc = w1e + (size_t)(kt + bk) * NF + nt * 128 + bn;
        __builtin_prefetch(bsrc + 32 * NF, 0, 0);   // one K-stage ahead
        bv.q[0] = *(const uint4*)bsrc;
        bv.q[1] = *(const uint4*)(bsrc + 8);
#pragma unroll
        for (int j = 0; j < 16; j++) Bs[buf][bn + j][bk] = bv.s[j];
    };

    const int S = DM / 32;
    stage(0, 0);
    for (int i = 0; i < S; i++) {
        const int buf = i & 1;
        if (i + 1 < S) { stage(buf ^ 1, (i + 1) * 32); wait_asynccnt<2>(); }
        else          { wait_asynccnt<0>(); }
        __syncthreads();

        Frag a[2], b[4];
        const uint32_t* As32 = (const uint32_t*)As[buf];
        const uint32_t* Bs32 = (const uint32_t*)Bs[buf];
#pragma unroll
        for (int sm = 0; sm < 2; sm++) {
            const uint32_t* p = As32 + (wm * 32 + sm * 16 + fm) * 16 + akk;
#pragma unroll
            for (int j = 0; j < 4; j++) { a[sm].u[j] = p[j]; a[sm].u[j + 4] = p[4 + j]; }
        }
#pragma unroll
        for (int sn = 0; sn < 4; sn++) {
            const uint32_t* p = Bs32 + (wn * 64 + sn * 16 + fm) * 16 + bkk;
#pragma unroll
            for (int j = 0; j < 8; j++) b[sn].u[j] = p[j];
        }
#pragma unroll
        for (int sm = 0; sm < 2; sm++)
#pragma unroll
            for (int sn = 0; sn < 4; sn++)
                acc[sm][sn] = __builtin_amdgcn_wmma_f32_16x16x32_bf16(
                    false, a[sm].v, false, b[sn].v, (short)0, acc[sm][sn], false, false);

        __syncthreads();
    }

    const int cm = (lane >> 4) * 8;
    const int cn = lane & 15;
#pragma unroll
    for (int sm = 0; sm < 2; sm++) {
#pragma unroll
        for (int v = 0; v < 8; v++) {
            const int r = mt * 128 + wm * 32 + sm * 16 + cm + v;
            if (r >= cnt) continue;
#pragma unroll
            for (int sn = 0; sn < 4; sn++) {
                const int f = nt * 128 + wn * 64 + sn * 16 + cn;
                float val = acc[sm][sn][v] + b1[e * NF + f];
                h[(size_t)r * NF + f] = f2bf(gelu_exact(val));
            }
        }
    }
}

// ---------------- GEMM2: out[tok] += gate * (h @ w2[e] + b2[e]) ----------------
__global__ __launch_bounds__(256) void moe_gemm2(
    const uint16_t* __restrict__ h, const uint16_t* __restrict__ wb2,
    const float* __restrict__ b2, const int* __restrict__ counts,
    const int* __restrict__ tok_ids, const float* __restrict__ gate_vals,
    float* __restrict__ out, int e)
{
    __shared__ __align__(16) uint16_t As[2][128][32];
    __shared__ __align__(16) uint16_t Bs[2][128][32];

    const int cnt = counts[e];
    const int mt = blockIdx.x, nt = blockIdx.y;
    if (mt * 128 >= cnt) return;

    const int tid  = threadIdx.x;
    const int lane = tid & 31, wid = tid >> 5;
    const int wm   = wid & 3,  wn  = wid >> 2;

    const uint16_t* w2e = wb2 + (size_t)e * NF * DM;

    const int arow = tid >> 1, acol = (tid & 1) * 16;
    const int r_load = mt * 128 + arow;
    const bool avalid = (r_load < cnt);
    const int bk = tid >> 3, bn = (tid & 7) * 16;

    const v8f vzero = {0,0,0,0,0,0,0,0};
    v8f acc[2][4];
#pragma unroll
    for (int i = 0; i < 2; i++)
#pragma unroll
        for (int j = 0; j < 4; j++) acc[i][j] = vzero;

    const int fm  = lane & 15;
    const int akk = (lane >> 4) * 8;
    const int bkk = (lane >> 4) * 8;

    auto stage = [&](int buf, int kt) {
        if (avalid) {
            const uint16_t* src = h + (size_t)r_load * NF + kt + acol;
            uint32_t l = lds_addr32(&As[buf][arow][acol]);
            async_b128(l, src);
            async_b128(l + 16, src + 8);
        } else {
            const uint4 z = make_uint4(0u, 0u, 0u, 0u);
            *(uint4*)&As[buf][arow][acol]     = z;
            *(uint4*)&As[buf][arow][acol + 8] = z;
        }
        union { uint4 q[2]; uint16_t s[16]; } bv;
        const uint16_t* bsrc = w2e + (size_t)(kt + bk) * DM + nt * 128 + bn;
        __builtin_prefetch(bsrc + 32 * DM, 0, 0);
        bv.q[0] = *(const uint4*)bsrc;
        bv.q[1] = *(const uint4*)(bsrc + 8);
#pragma unroll
        for (int j = 0; j < 16; j++) Bs[buf][bn + j][bk] = bv.s[j];
    };

    const int S = NF / 32;
    stage(0, 0);
    for (int i = 0; i < S; i++) {
        const int buf = i & 1;
        if (i + 1 < S) { stage(buf ^ 1, (i + 1) * 32); wait_asynccnt<2>(); }
        else          { wait_asynccnt<0>(); }
        __syncthreads();

        Frag a[2], b[4];
        const uint32_t* As32 = (const uint32_t*)As[buf];
        const uint32_t* Bs32 = (const uint32_t*)Bs[buf];
#pragma unroll
        for (int sm = 0; sm < 2; sm++) {
            const uint32_t* p = As32 + (wm * 32 + sm * 16 + fm) * 16 + akk;
#pragma unroll
            for (int j = 0; j < 4; j++) { a[sm].u[j] = p[j]; a[sm].u[j + 4] = p[4 + j]; }
        }
#pragma unroll
        for (int sn = 0; sn < 4; sn++) {
            const uint32_t* p = Bs32 + (wn * 64 + sn * 16 + fm) * 16 + bkk;
#pragma unroll
            for (int j = 0; j < 8; j++) b[sn].u[j] = p[j];
        }
#pragma unroll
        for (int sm = 0; sm < 2; sm++)
#pragma unroll
            for (int sn = 0; sn < 4; sn++)
                acc[sm][sn] = __builtin_amdgcn_wmma_f32_16x16x32_bf16(
                    false, a[sm].v, false, b[sn].v, (short)0, acc[sm][sn], false, false);

        __syncthreads();
    }

    const int cm = (lane >> 4) * 8;
    const int cn = lane & 15;
#pragma unroll
    for (int sm = 0; sm < 2; sm++) {
#pragma unroll
        for (int v = 0; v < 8; v++) {
            const int r = mt * 128 + wm * 32 + sm * 16 + cm + v;
            if (r >= cnt) continue;
            const int   tok = tok_ids[e * NTOK + r];
            const float g   = gate_vals[e * NTOK + r];
#pragma unroll
            for (int sn = 0; sn < 4; sn++) {
                const int d = nt * 128 + wn * 64 + sn * 16 + cn;
                float val = g * (acc[sm][sn][v] + b2[e * DM + d]);
                atomicAdd(&out[(size_t)tok * DM + d], val);
            }
        }
    }
}

extern "C" void kernel_launch(void* const* d_in, const int* in_sizes, int n_in,
                              void* d_out, int out_size, void* d_ws, size_t ws_size,
                              hipStream_t stream) {
    (void)in_sizes; (void)n_in; (void)out_size; (void)ws_size;
    const float* x  = (const float*)d_in[0];
    const float* gw = (const float*)d_in[1];
    const float* w1 = (const float*)d_in[2];
    const float* b1 = (const float*)d_in[3];
    const float* w2 = (const float*)d_in[4];
    const float* b2 = (const float*)d_in[5];
    float* out = (float*)d_out;

    char* ws = (char*)d_ws;
    size_t off = 0;
    auto take = [&](size_t bytes) -> char* {
        char* p = ws + off;
        off = (off + bytes + 255) & ~(size_t)255;
        return p;
    };
    uint16_t* xb      = (uint16_t*)take((size_t)NTOK * DM * 2);
    uint16_t* wb1     = (uint16_t*)take((size_t)NE * DM * NF * 2);
    uint16_t* wb2     = (uint16_t*)take((size_t)NE * NF * DM * 2);
    uint16_t* hbuf    = (uint16_t*)take((size_t)NTOK * NF * 2);
    int*      counts  = (int*)take(NE * sizeof(int));
    int*      tok_ids = (int*)take((size_t)NE * NTOK * sizeof(int));
    float*    gates   = (float*)take((size_t)NE * NTOK * sizeof(float));

    moe_init<<<dim3((NTOK * DM) / 256), 256, 0, stream>>>(out, NTOK * DM, counts);
    cvt_bf16v<<<dim3((NTOK * DM) / (256 * 8)), 256, 0, stream>>>(x, xb, NTOK * DM);
    cvt_bf16v<<<dim3((NE * DM * NF) / (256 * 8)), 256, 0, stream>>>(w1, wb1, NE * DM * NF);
    cvt_bf16v<<<dim3((NE * NF * DM) / (256 * 8)), 256, 0, stream>>>(w2, wb2, NE * NF * DM);
    moe_gate<<<dim3(NTOK / 8), 256, 0, stream>>>(x, gw, counts, tok_ids, gates);

    for (int e = 0; e < NE; e++) {
        moe_gemm1<<<dim3(64, NF / 128), 256, 0, stream>>>(xb, wb1, b1, counts, tok_ids, hbuf, e);
        moe_gemm2<<<dim3(64, DM / 128), 256, 0, stream>>>(hbuf, wb2, b2, counts, tok_ids, gates, out, e);
    }
}
